// MaskedLSTMCell_56178172232433
// MI455X (gfx1250) — compile-verified, matrix operand order checked by reading, not run on hardware
//
#include <hip/hip_runtime.h>

// ---------------------------------------------------------------------------
// Masked LSTM for MI455X (gfx1250, wave32, WMMA + TDM)
//   T=256, B=64, I=1024, H=1024, gates = [i, f, o, c]
//   1) Convert X, (Wx*mask_x), (Wh*mask_h) to bf16 (RNE).
//   2) One big WMMA GEMM: Xg[TB x 4H] = Xbf @ Wx^T + bias  (f32 accum).
//   3) 256 sequential step kernels: Hprev staged to LDS via tensor_load_to_lds
//      (TDM, padded layout), G = Hprev @ Wh^T + Xg[t], fused LSTM cell.
// ---------------------------------------------------------------------------

typedef __attribute__((ext_vector_type(16))) __bf16 v16bf;
typedef __attribute__((ext_vector_type(8)))  float  v8f;
typedef __attribute__((ext_vector_type(4)))  unsigned u32x4;
typedef __attribute__((ext_vector_type(8)))  int      i32x8;
typedef __attribute__((ext_vector_type(4)))  int      i32x4;

typedef unsigned short bf16s;   // raw bf16 storage

#define T_STEPS 256
#define BATCH   64
#define IDIM    1024
#define HDIM    1024
#define NG      4096            // 4 * HDIM (gate-major columns: g*1024 + h)
#define ROWS    16384           // T * B

// LDS layout for lstm_step: TDM-padded A buffer + gate staging buffer.
// Row = 512 dwords + 4-dword pad every 256 dwords -> 520 dwords = 2080 B.
#define ABUF_BYTES (BATCH * 520 * 4)        // 133120
#define GBUF_BYTES (4 * BATCH * 32 * 4)     // 32768
#define STEP_LDS_BYTES (ABUF_BYTES + GBUF_BYTES)

// ---- bf16 helpers ----------------------------------------------------------
__device__ __forceinline__ unsigned short f2bf_bits(float f) {
  unsigned u = __builtin_bit_cast(unsigned, f);
  unsigned r = u + 0x7fffu + ((u >> 16) & 1u);   // round-to-nearest-even
  return (unsigned short)(r >> 16);
}

__device__ __forceinline__ float fast_sigmoid(float x) {
  return 1.0f / (1.0f + __expf(-x));             // v_exp_f32 + v_rcp_f32
}
__device__ __forceinline__ float fast_tanh(float x) {
  float e = __expf(2.0f * x);
  return 1.0f - 2.0f / (e + 1.0f);               // saturates correctly at +-1
}

// ---- WMMA bf16 fragment loader (global) ------------------------------------
// 16-bit A-matrix 16x32 layout (ISA 7.12.2): lane L<16 holds row M=L,
// k = {kb..kb+7, kb+16..kb+23}; lane L>=16 holds row M=L-16,
// k = {kb+8..kb+15, kb+24..kb+31}. Rows contiguous in k -> two b128 loads.
__device__ __forceinline__ v16bf load_frag(const bf16s* __restrict__ base,
                                           int stride, int rc0, int kb) {
  int lane = threadIdx.x & 31;
  int half = lane >> 4;
  int rc   = rc0 + (lane & 15);
  const bf16s* p = base + (size_t)rc * stride + kb + half * 8;
  union { uint4 u[2]; v16bf v; } f;
  f.u[0] = *(const uint4*)(p);
  f.u[1] = *(const uint4*)(p + 16);
  return f.v;
}

// Byte offset in the TDM-padded LDS A buffer for dword index d
// (pad_interval = 256 dwords, pad_amount = 4 dwords -> +16 B per 256 dwords).
__device__ __forceinline__ int lds_pad_off(int d) {
  return 4 * d + 16 * (d >> 8);
}

// A-fragment from padded LDS buffer (ds_load_b128 x2).
__device__ __forceinline__ v16bf load_frag_lds(const char* Ab, int m0, int kb) {
  int lane = threadIdx.x & 31;
  int half = lane >> 4;
  int m    = m0 + (lane & 15);
  int kd   = (kb >> 1) + half * 4;        // dword offset within row
  int d0   = m * 512 + kd;
  int d1   = d0 + 8;                      // +16 elements
  union { uint4 u[2]; v16bf v; } f;
  f.u[0] = *(const uint4*)(Ab + lds_pad_off(d0));
  f.u[1] = *(const uint4*)(Ab + lds_pad_off(d1));
  return f.v;
}

__device__ __forceinline__ v8f wmma_bf16(v16bf a, v16bf b, v8f c) {
  return __builtin_amdgcn_wmma_f32_16x16x32_bf16(
      /*neg_a=*/false, a, /*neg_b=*/false, b,
      /*c_mod=*/(short)0, c, /*reuse_a=*/false, /*reuse_b=*/false);
}

// ---- Kernel 1: fp32 (-> * mask) -> bf16, 4 elements / thread ---------------
__global__ void lstm_conv_bf16(const float* __restrict__ src,
                               const float* __restrict__ mask,
                               bf16s* __restrict__ dst, int n4) {
  int i = blockIdx.x * blockDim.x + threadIdx.x;
  if (i >= n4) return;
  float4 s = ((const float4*)src)[i];
  if (mask) {
    float4 m = ((const float4*)mask)[i];
    s.x *= m.x; s.y *= m.y; s.z *= m.z; s.w *= m.w;
  }
  ushort4 o;
  o.x = f2bf_bits(s.x); o.y = f2bf_bits(s.y);
  o.z = f2bf_bits(s.z); o.w = f2bf_bits(s.w);
  ((ushort4*)dst)[i] = o;
}

// ---- Kernel 2: init recurrent state ----------------------------------------
__global__ void lstm_init_state(const float* __restrict__ H0,
                                const float* __restrict__ C0,
                                bf16s* __restrict__ Hbf,
                                float* __restrict__ Cst, int n) {
  int i = blockIdx.x * blockDim.x + threadIdx.x;
  if (i >= n) return;
  Hbf[i] = f2bf_bits(H0[i]);
  Cst[i] = C0[i];
}

// ---- Kernel 3: big input GEMM  Xg = Xbf @ Wx^T + bias ----------------------
// Block = 256 thr = 8 waves (2 M x 4 N), wave tile 32x32, block tile 64x128.
__global__ void __launch_bounds__(256)
lstm_gemm_x(const bf16s* __restrict__ A,     // ROWS x IDIM
            const bf16s* __restrict__ Bm,    // NG rows x IDIM (Wx masked bf16)
            const float* __restrict__ bias,  // NG (g*1024+h ordering)
            float* __restrict__ Cout) {      // ROWS x NG
  int wave = threadIdx.x >> 5;
  int lane = threadIdx.x & 31;
  int m0 = blockIdx.y * 64  + (wave >> 2) * 32;
  int n0 = blockIdx.x * 128 + (wave & 3)  * 32;

  v8f acc00 = {}, acc01 = {}, acc10 = {}, acc11 = {};
  for (int kb = 0; kb < IDIM; kb += 32) {
    v16bf a0 = load_frag(A,  IDIM, m0,      kb);
    v16bf a1 = load_frag(A,  IDIM, m0 + 16, kb);
    v16bf b0 = load_frag(Bm, IDIM, n0,      kb);
    v16bf b1 = load_frag(Bm, IDIM, n0 + 16, kb);
    acc00 = wmma_bf16(a0, b0, acc00);
    acc01 = wmma_bf16(a0, b1, acc01);
    acc10 = wmma_bf16(a1, b0, acc10);
    acc11 = wmma_bf16(a1, b1, acc11);
  }

  int half = lane >> 4, nl = lane & 15;
  v8f accs[2][2] = {{acc00, acc01}, {acc10, acc11}};
#pragma unroll
  for (int i = 0; i < 2; ++i)
#pragma unroll
    for (int j = 0; j < 2; ++j) {
      int n = n0 + j * 16 + nl;
      float bv = bias[n];
#pragma unroll
      for (int r = 0; r < 8; ++r) {
        int m = m0 + i * 16 + r + 8 * half;       // C/D layout (ISA 7.12.2)
        Cout[(size_t)m * NG + n] = accs[i][j][r] + bv;
      }
    }
}

// ---- Kernel 4: one recurrent LSTM step -------------------------------------
// Grid = 32 blocks; block owns a 32-wide h-slice across ALL 4 gates.
// Hprev (64x1024 bf16, 128 KB) is staged into padded LDS by the Tensor Data
// Mover (one DMA per workgroup), then A-fragments come from ds_load_b128.
__global__ void __launch_bounds__(256)
lstm_step(int t,
          const bf16s* __restrict__ Hprev,   // 64 x 1024 bf16
          const bf16s* __restrict__ Whb,     // 4 x 1024 x 1024 bf16 (masked)
          const float* __restrict__ Xg,      // ROWS x NG (precomputed + bias)
          float* __restrict__ Cst,           // 64 x 1024 fp32 state (in place)
          bf16s* __restrict__ Hnext,         // 64 x 1024 bf16
          float* __restrict__ out) {         // T x 64 x 1024 fp32
  extern __shared__ char smem[];
  char*  Ab = smem;                          // padded bf16 A buffer (133120 B)
  float* gb = (float*)(smem + ABUF_BYTES);   // gate staging (4*64*32 f32)

  int wave = threadIdx.x >> 5;
  int lane = threadIdx.x & 31;
  int hbase = blockIdx.x * 32;
  int g   = wave >> 1;
  int nl0 = (wave & 1) * 16;

  // ---- Stage Hprev -> LDS via TDM (wave 0 issues one DMA) ----
#if __has_builtin(__builtin_amdgcn_tensor_load_to_lds)
  if (wave == 0) {
    unsigned long long ga = (unsigned long long)Hprev;
    unsigned lds_addr = (unsigned)(unsigned long long)(void*)Ab;
    // D# group 0: count=1, lds_addr, global_addr, type=2 ("image")
    u32x4 g0;
    g0[0] = 1u;                                   // count=1, is_restore=0
    g0[1] = lds_addr;                             // LDS byte address
    g0[2] = (unsigned)(ga & 0xffffffffull);       // global_addr[31:0]
    g0[3] = (unsigned)((ga >> 32) & 0x1ffffffull) // global_addr[56:32]
            | (2u << 30);                         // type=2
    // D# group 1: data_size=2B, pad 4 dwords every 256 dwords,
    // tensor 1024x64, tile 1024x64, dim0 stride 1024.
    i32x8 g1;
    g1[0] = (int)((1u << 16)        // data_size = 2 bytes
                | (1u << 20)        // pad_enable
                | (7u << 22)        // pad_interval = 256 dwords
                | (3u << 25));      // pad_amount   = 4 dwords
    g1[1] = (int)((1024u & 0xffffu) << 16);          // tensor_dim0[15:0]
    g1[2] = (int)((1024u >> 16) | (64u << 16));      // dim0[31:16], dim1[15:0]
    g1[3] = (int)((64u >> 16) | (1024u << 16));      // dim1[31:16], tile_dim0
    g1[4] = (int)(64u);                              // tile_dim1=64, tile_dim2=0
    g1[5] = (int)1024u;                              // tensor_dim0_stride[31:0]
    g1[6] = 0;                                       // stride0[47:32], stride1 lo
    g1[7] = 0;                                       // stride1 hi
    i32x4 gz4  = {0, 0, 0, 0};                       // groups 2/3 unused (2-D)
    i32x8 gz8  = {0, 0, 0, 0, 0, 0, 0, 0};           // extra group (unused)
    __builtin_amdgcn_tensor_load_to_lds(g0, g1, gz4, gz4, gz8, 0);
#if __has_builtin(__builtin_amdgcn_s_wait_tensorcnt)
    __builtin_amdgcn_s_wait_tensorcnt(0);
#else
    asm volatile("s_wait_tensorcnt 0x0" ::: "memory");
#endif
  }
#else
  // Fallback: cooperative copy into the same padded layout.
  for (int c = threadIdx.x; c < BATCH * 128; c += 256) {   // 16-B chunks
    int d = c * 4;                                          // dword index
    int e = d * 2;                                          // element index
    int m = e >> 10, k = e & 1023;
    *(uint4*)(Ab + lds_pad_off(d)) = *(const uint4*)(Hprev + m * 1024 + k);
  }
#endif
  __syncthreads();

  const bf16s* Brow = Whb + (size_t)g * HDIM * HDIM;

  v8f acc[4] = {};
  for (int kb = 0; kb < HDIM; kb += 32) {
    v16bf bf = load_frag(Brow, HDIM, hbase + nl0, kb);
#pragma unroll
    for (int mt = 0; mt < 4; ++mt) {
      v16bf af = load_frag_lds(Ab, mt * 16, kb);
      acc[mt] = wmma_bf16(af, bf, acc[mt]);
    }
  }

  int half = lane >> 4;
  int nl = nl0 + (lane & 15);                 // local col within 32-slice
  int hcol = hbase + nl;                      // hidden index
  const float* Xrow = Xg + (size_t)t * BATCH * NG;
#pragma unroll
  for (int mt = 0; mt < 4; ++mt)
#pragma unroll
    for (int r = 0; r < 8; ++r) {
      int m = mt * 16 + r + 8 * half;         // batch row
      gb[((g * BATCH) + m) * 32 + nl] = acc[mt][r]
          + Xrow[(size_t)m * NG + g * HDIM + hcol];
    }
  __syncthreads();

  // Fused elementwise LSTM cell: gates 0=i, 1=f, 2=o, 3=c
  for (int e = threadIdx.x; e < BATCH * 32; e += 256) {
    int m  = e >> 5;
    int hl = e & 31;
    int h  = hbase + hl;
    float gi = fast_sigmoid(gb[(0 * BATCH + m) * 32 + hl]);
    float gf = fast_sigmoid(gb[(1 * BATCH + m) * 32 + hl]);
    float go = fast_sigmoid(gb[(2 * BATCH + m) * 32 + hl]);
    float gc = fast_tanh   (gb[(3 * BATCH + m) * 32 + hl]);
    size_t sidx = (size_t)m * HDIM + h;
    float cn = gf * Cst[sidx] + gi * gc;
    float hn = go * fast_tanh(cn);
    Cst[sidx] = cn;
    out[(size_t)t * BATCH * HDIM + sidx] = hn;
    Hnext[sidx] = f2bf_bits(hn);
  }
}

// ---------------------------------------------------------------------------
extern "C" void kernel_launch(void* const* d_in, const int* in_sizes, int n_in,
                              void* d_out, int out_size, void* d_ws, size_t ws_size,
                              hipStream_t stream) {
  const float* X    = (const float*)d_in[0];  // (T,B,I)
  const float* Wx   = (const float*)d_in[1];  // (4,H,I)
  const float* Wh   = (const float*)d_in[2];  // (4,H,H)
  const float* bias = (const float*)d_in[3];  // (4,H)
  const float* mx   = (const float*)d_in[4];  // (4,H,I)
  const float* mh   = (const float*)d_in[5];  // (4,H,H)
  const float* H0   = (const float*)d_in[6];  // (B,H)
  const float* C0   = (const float*)d_in[7];  // (B,H)
  float* out = (float*)d_out;                 // (T,B,H)

  // Workspace partition (~305 MB)
  char* ws = (char*)d_ws;
  size_t off = 0;
  float* Xg  = (float*)(ws + off); off += (size_t)ROWS * NG * sizeof(float);
  bf16s* Xbf = (bf16s*)(ws + off); off += (size_t)ROWS * IDIM * sizeof(bf16s);
  bf16s* Wxb = (bf16s*)(ws + off); off += (size_t)4 * HDIM * IDIM * sizeof(bf16s);
  bf16s* Whb = (bf16s*)(ws + off); off += (size_t)4 * HDIM * HDIM * sizeof(bf16s);
  bf16s* Hbf0 = (bf16s*)(ws + off); off += (size_t)BATCH * HDIM * sizeof(bf16s);
  bf16s* Hbf1 = (bf16s*)(ws + off); off += (size_t)BATCH * HDIM * sizeof(bf16s);
  float* Cst  = (float*)(ws + off); off += (size_t)BATCH * HDIM * sizeof(float);
  (void)ws_size; (void)in_sizes; (void)n_in; (void)out_size;

  // 1) bf16 conversions (mask applied to weights once)
  {
    int n4 = ROWS * IDIM / 4;
    lstm_conv_bf16<<<(n4 + 255) / 256, 256, 0, stream>>>(X, nullptr, Xbf, n4);
  }
  {
    int n4 = 4 * HDIM * IDIM / 4;
    lstm_conv_bf16<<<(n4 + 255) / 256, 256, 0, stream>>>(Wx, mx, Wxb, n4);
  }
  {
    int n4 = 4 * HDIM * HDIM / 4;
    lstm_conv_bf16<<<(n4 + 255) / 256, 256, 0, stream>>>(Wh, mh, Whb, n4);
  }
  {
    int n = BATCH * HDIM;
    lstm_init_state<<<(n + 255) / 256, 256, 0, stream>>>(H0, C0, Hbf0, Cst, n);
  }

  // 2) big input-side GEMM with bias fused
  {
    dim3 grid(NG / 128, ROWS / 64);   // (32, 256)
    lstm_gemm_x<<<grid, 256, 0, stream>>>(Xbf, Wxb, bias, Xg);
  }

  // 3) sequential recurrence (stream order = inter-step sync)
  for (int t = 0; t < T_STEPS; ++t) {
    const bf16s* hp = (t & 1) ? Hbf1 : Hbf0;
    bf16s*       hn = (t & 1) ? Hbf0 : Hbf1;
    lstm_step<<<HDIM / 32, 256, STEP_LDS_BYTES, stream>>>(
        t, hp, Whb, Xg, Cst, hn, out);
  }
}
